// RNNX_64278480552528
// MI455X (gfx1250) — compile-verified, matrix operand order checked by reading, not run on hardware
//
#include <hip/hip_runtime.h>

typedef _Float16 v16h __attribute__((ext_vector_type(16)));
typedef float    v8f  __attribute__((ext_vector_type(8)));

namespace {
constexpr int    S_STEPS  = 256;
constexpr int    BATCH    = 64;
constexpr int    HID      = 2048;
constexpr int    KDIM     = 4096;              // I + H == H + H
constexpr int    GRID     = 64;                // 32 N-panels x 2 M-halves
constexpr int    TPB      = 256;               // 8 waves (wave32)
constexpr int    NTHREADS = GRID * TPB;
constexpr int    KSLICE   = KDIM / 8;          // 512 K per wave
constexpr int    NCHUNK   = KSLICE / 32;       // 16 chunks of K=32
constexpr size_t W_ELEMS  = (size_t)HID * KDIM;
constexpr size_t H_ELEMS  = (size_t)BATCH * HID;
constexpr size_t X_ELEMS  = (size_t)S_STEPS * H_ELEMS;

// async LDS ring: per wave, D slots of (B 4KB + A 2KB)
constexpr int DRING      = 3;
constexpr int SLOT_A_OFF = 4096;
constexpr int SLOT_BYTES = 6144;
constexpr int RING_BYTES = DRING * SLOT_BYTES;     // 18 KB / wave
constexpr int LDS_BYTES  = 8 * RING_BYTES;         // 144 KB

union V16 { v16h v; _Float16 h[16]; uint4 q[2]; };
union H4  { _Float16 h[4]; uint2 q; };
} // namespace

// ---- CDNA5 async global->LDS copy, GVS mode: mem = SGPR64 + VGPR32 (ASYNCcnt) ----
__device__ __forceinline__ void async_b128_gvs(unsigned lds_addr, unsigned voff,
                                               unsigned long long sbase) {
    asm volatile("global_load_async_to_lds_b128 %0, %1, %2"
                 :: "v"(lds_addr), "v"(voff), "s"(sbase) : "memory");
}

template <int N>
__device__ __forceinline__ void wait_async() {
#if __has_builtin(__builtin_amdgcn_s_wait_asynccnt)
    __builtin_amdgcn_s_wait_asynccnt(N);
#else
    if constexpr (N == 24)      asm volatile("s_wait_asynccnt 0x18" ::: "memory");
    else if constexpr (N == 12) asm volatile("s_wait_asynccnt 0xc"  ::: "memory");
    else                        asm volatile("s_wait_asynccnt 0x0"  ::: "memory");
#endif
}

// ---- per-wave producer wait / per-phase signal (counters memset to 0 per call) ----
__device__ __forceinline__ void wave_wait(unsigned* __restrict__ cnt, int phase) {
    while (__hip_atomic_load(&cnt[phase], __ATOMIC_ACQUIRE, __HIP_MEMORY_SCOPE_AGENT)
           < (unsigned)GRID) {
        __builtin_amdgcn_s_sleep(1);
    }
}

__device__ __forceinline__ void phase_signal(unsigned* __restrict__ cnt, int phase) {
    __threadfence();
    __syncthreads();
    if (threadIdx.x == 0) {
        __hip_atomic_fetch_add(&cnt[phase], 1u, __ATOMIC_RELEASE, __HIP_MEMORY_SCOPE_AGENT);
    }
}

// ---------------- one layer GEMM phase: D = tanh(A @ W^T + b) -----------------
// A = [alo (first 2048 cols)] ++ [ahi (second 2048 cols)], all f16; K split 8 ways.
__device__ __forceinline__ void gemm_phase(
    const _Float16* alo,                   // f16 first-half A source
    const _Float16* ahi,                   // f16 second-half A source (h_prev)
    int out_off,                           // element offset into statesw
    const _Float16* __restrict__ Wh,       // [HID][KDIM] f16
    const float*    __restrict__ bias,     // [HID] fp32
    _Float16*       __restrict__ statesw,
    float*          __restrict__ outbase,
    long long o0, long long o1,            // fp32 output offsets, -1 = skip
    char* smem,                            // LDS: async ring (reduction aliases base)
    unsigned* __restrict__ cnt,
    int wait_lo, int wait_hi, int sig)
{
    const int tid    = threadIdx.x;
    const int lane   = tid & 31;
    const int kq     = lane >> 4;
    const int nlo    = lane & 15;
    const int npanel = blockIdx.x >> 1;
    const int mhalf  = blockIdx.x & 1;
    const int n0     = npanel * 64;
    const int arow   = mhalf * 32;

    const int  wv      = __builtin_amdgcn_readfirstlane(tid >> 5);
    const int  kbase   = wv * KSLICE;
    const bool lo_half = (kbase < HID);
    const int  kcol0   = lo_half ? kbase : (kbase - HID);

    // scalar 64-bit bases (wave-uniform): bumped by 64B per K-chunk
    const unsigned long long wb0 =
        (unsigned long long)(const void*)(Wh + kbase);
    const unsigned long long ab0 =
        (unsigned long long)(const void*)((lo_half ? alo : ahi) + kcol0);

    // loop-invariant 32-bit per-lane global offsets (bytes)
    unsigned voffB[8], voffA[4];
#pragma unroll
    for (int nt = 0; nt < 4; ++nt)
#pragma unroll
        for (int h = 0; h < 2; ++h)
            voffB[nt * 2 + h] =
                (unsigned)(((n0 + nlo + nt * 16) * KDIM + kq * 8 + h * 16) * 2);
#pragma unroll
    for (int mt = 0; mt < 2; ++mt)
#pragma unroll
        for (int h = 0; h < 2; ++h)
            voffA[mt * 2 + h] =
                (unsigned)(((arow + nlo + mt * 16) * HID + kq * 8 + h * 16) * 2);

    // loop-invariant LDS destination table (3 slots x 12 fragments)
    char* ring = smem + wv * RING_BYTES;
    const unsigned ring_u = (unsigned)(size_t)ring + (unsigned)lane * 16;
    unsigned ldsB[DRING][8], ldsA[DRING][4];
#pragma unroll
    for (int s = 0; s < DRING; ++s) {
#pragma unroll
        for (int j = 0; j < 8; ++j) ldsB[s][j] = ring_u + s * SLOT_BYTES + j * 512;
#pragma unroll
        for (int j = 0; j < 4; ++j) ldsA[s][j] = ring_u + s * SLOT_BYTES + SLOT_A_OFF + j * 512;
    }

    auto issue_B = [&](int c) {
        const unsigned long long sb = wb0 + (unsigned long long)(c * 64);
        const int s = c % DRING;
#pragma unroll
        for (int j = 0; j < 8; ++j) async_b128_gvs(ldsB[s][j], voffB[j], sb);
    };
    auto issue_A = [&](int c) {
        const unsigned long long sb = ab0 + (unsigned long long)(c * 64);
        const int s = c % DRING;
#pragma unroll
        for (int j = 0; j < 4; ++j) async_b128_gvs(ldsA[s][j], voffA[j], sb);
    };

    v8f acc[2][4];
#pragma unroll
    for (int mt = 0; mt < 2; ++mt)
#pragma unroll
        for (int nt = 0; nt < 4; ++nt)
            acc[mt][nt] = v8f{0.f, 0.f, 0.f, 0.f, 0.f, 0.f, 0.f, 0.f};

    // ---- prologue: B(0) is producer-independent; A needs the producer done ----
    issue_B(0);
    const int wp_phase = lo_half ? wait_lo : wait_hi;
    if (wp_phase >= 0) wave_wait(cnt, wp_phase);
    issue_A(0);
    issue_B(1); issue_A(1);
    issue_B(2); issue_A(2);

    // ---- K loop: async ring keeps DRING chunks in flight; no barriers ----
#pragma unroll
    for (int c = 0; c < NCHUNK; ++c) {
        if (c < NCHUNK - 2)       wait_async<(DRING - 1) * 12>();   // 24
        else if (c == NCHUNK - 2) wait_async<12>();
        else                      wait_async<0>();
        __builtin_amdgcn_sched_barrier(0);

        const char* sp = ring + (c % DRING) * SLOT_BYTES + lane * 16;
        V16 b[4], a[2];
#pragma unroll
        for (int mt = 0; mt < 2; ++mt)              // A first: first WMMA ready sooner
#pragma unroll
            for (int h = 0; h < 2; ++h)
                a[mt].q[h] = *(const uint4*)(sp + SLOT_A_OFF + (mt * 2 + h) * 512);
#pragma unroll
        for (int nt = 0; nt < 4; ++nt)
#pragma unroll
            for (int h = 0; h < 2; ++h)
                b[nt].q[h] = *(const uint4*)(sp + (nt * 2 + h) * 512);
        __builtin_amdgcn_sched_barrier(0);

#pragma unroll
        for (int nt = 0; nt < 4; ++nt) {
            acc[0][nt] = __builtin_amdgcn_wmma_f32_16x16x32_f16(false, a[0].v, false,
                             b[nt].v, (short)0, acc[0][nt], false, false);
            acc[1][nt] = __builtin_amdgcn_wmma_f32_16x16x32_f16(false, a[1].v, false,
                             b[nt].v, (short)0, acc[1][nt], false, false);
        }
        __builtin_amdgcn_sched_barrier(0);

        if (c + DRING < NCHUNK) {       // refill freed slot (ds reads already complete)
            issue_B(c + DRING);
            issue_A(c + DRING);
        }
        __builtin_amdgcn_sched_barrier(0);
    }

    // ---- cross-wave reduction (red aliases ring base; ring is drained) ----
    float* red = (float*)smem;
    __syncthreads();                       // everyone done consuming the ring
    if (wv >= 4) {
#pragma unroll
        for (int mt = 0; mt < 2; ++mt)
#pragma unroll
            for (int nt = 0; nt < 4; ++nt) {
                float* d = red + (wv - 4) * 2048 + (nt * 2 + mt) * 256 + lane * 8;
                *(v8f*)d = acc[mt][nt];
            }
    }
    __syncthreads();
    if (wv < 4) {
#pragma unroll
        for (int mt = 0; mt < 2; ++mt)
#pragma unroll
            for (int nt = 0; nt < 4; ++nt) {
                float* d = red + wv * 2048 + (nt * 2 + mt) * 256 + lane * 8;
                *(v8f*)d = acc[mt][nt] + *(const v8f*)d;
            }
    }
    __syncthreads();

    // ---- final sum + bias + tanh; thread <-> (tile = wave, lane) ----
    {
        const int t  = wv;
        const int mt = t & 1, nt = t >> 1;
        const float* s0 = red + t * 256 + lane * 8;
        v8f sum = *(const v8f*)(s0) + *(const v8f*)(s0 + 2048)
                + *(const v8f*)(s0 + 4096) + *(const v8f*)(s0 + 6144);
        const int   nn = n0 + nt * 16 + nlo;
        const float bv = bias[nn];
#pragma unroll
        for (int r = 0; r < 8; ++r) {
            const int    m   = mhalf * 32 + mt * 16 + kq * 8 + r;
            const float  v   = tanhf(sum[r] + bv);
            const size_t idx = (size_t)m * HID + nn;
            statesw[out_off + idx] = (_Float16)v;
            if (o0 >= 0) outbase[(size_t)o0 + idx] = v;
            if (o1 >= 0) outbase[(size_t)o1 + idx] = v;
        }
    }

    phase_signal(cnt, sig);
}

// ---------------- persistent kernel: init + 256 steps x 2 layers ----------------
__global__ void __launch_bounds__(TPB, 1)
rnnx_persistent(const float* __restrict__ Xt,
                const float* __restrict__ W0, const float* __restrict__ b0,
                const float* __restrict__ W1, const float* __restrict__ b1,
                float* __restrict__ outbase,
                _Float16* __restrict__ W0h, _Float16* __restrict__ W1h,
                _Float16* __restrict__ Xh,        // f16 copy of Xt
                _Float16* __restrict__ hstates,   // 4 buffers of H_ELEMS halves
                unsigned* __restrict__ counters)
{
    __shared__ __align__(128) char smem[LDS_BYTES];   // 144 KB async ring (+aliased red)
    const unsigned gid = blockIdx.x * TPB + threadIdx.x;

    // ---- one-time per call: weights + X fp32 -> f16, zero h states ----
    for (size_t i = gid; i < W_ELEMS / 4; i += NTHREADS) {
        float4 w = ((const float4*)W0)[i];
        H4 t;
        t.h[0] = (_Float16)w.x; t.h[1] = (_Float16)w.y;
        t.h[2] = (_Float16)w.z; t.h[3] = (_Float16)w.w;
        ((uint2*)W0h)[i] = t.q;
    }
    for (size_t i = gid; i < W_ELEMS / 4; i += NTHREADS) {
        float4 w = ((const float4*)W1)[i];
        H4 t;
        t.h[0] = (_Float16)w.x; t.h[1] = (_Float16)w.y;
        t.h[2] = (_Float16)w.z; t.h[3] = (_Float16)w.w;
        ((uint2*)W1h)[i] = t.q;
    }
    for (size_t i = gid; i < X_ELEMS / 4; i += NTHREADS) {
        float4 w = ((const float4*)Xt)[i];
        H4 t;
        t.h[0] = (_Float16)w.x; t.h[1] = (_Float16)w.y;
        t.h[2] = (_Float16)w.z; t.h[3] = (_Float16)w.w;
        ((uint2*)Xh)[i] = t.q;
    }
    for (size_t i = gid; i < (4 * H_ELEMS) / 8; i += NTHREADS) {
        ((uint4*)hstates)[i] = make_uint4(0u, 0u, 0u, 0u);
    }
    phase_signal(counters, 0);
    wave_wait(counters, 0);

    const long long hT0_off = (long long)S_STEPS * H_ELEMS;
    const long long hT1_off = hT0_off + (long long)H_ELEMS;

    for (int s = 0; s < S_STEPS; ++s) {
        const int   p    = s & 1;
        const bool  last = (s == S_STEPS - 1);

        // layer 0: h0' = tanh([x | h0] @ W0^T + b0)   signal -> 1+2s
        gemm_phase(Xh + (size_t)s * H_ELEMS, hstates + (size_t)p * H_ELEMS,
                   /*out_off=*/(p ^ 1) * (int)H_ELEMS,
                   W0h, b0, hstates, outbase,
                   last ? hT0_off : -1, -1, smem, counters,
                   /*wait_lo=*/-1, /*wait_hi=*/(s > 0) ? (2 * s - 1) : -1,
                   /*sig=*/1 + 2 * s);

        // layer 1: h1' = tanh([h0' | h1] @ W1^T + b1)  signal -> 2+2s
        gemm_phase(hstates + (size_t)(p ^ 1) * H_ELEMS, hstates + (size_t)(2 + p) * H_ELEMS,
                   /*out_off=*/(2 + (p ^ 1)) * (int)H_ELEMS,
                   W1h, b1, hstates, outbase,
                   (long long)s * H_ELEMS, last ? hT1_off : -1, smem, counters,
                   /*wait_lo=*/1 + 2 * s, /*wait_hi=*/(s > 0) ? (2 * s) : -1,
                   /*sig=*/2 + 2 * s);
    }
}

// ---------------------------------- launch ----------------------------------
extern "C" void kernel_launch(void* const* d_in, const int* in_sizes, int n_in,
                              void* d_out, int out_size, void* d_ws, size_t ws_size,
                              hipStream_t stream) {
    const float* Xt  = (const float*)d_in[0];
    const float* W0  = (const float*)d_in[1];
    const float* b0v = (const float*)d_in[2];
    const float* W1  = (const float*)d_in[3];
    const float* b1v = (const float*)d_in[4];

    float* out = (float*)d_out;

    // workspace: [counters 4KB][W0h 16MB][W1h 16MB][Xh 64MB][hstates 1MB]
    char*      ws       = (char*)d_ws;
    unsigned*  counters = (unsigned*)ws;
    _Float16*  W0h      = (_Float16*)(ws + 4096);
    _Float16*  W1h      = W0h + W_ELEMS;
    _Float16*  Xh       = W1h + W_ELEMS;
    _Float16*  hstates  = Xh + X_ELEMS;

    hipMemsetAsync(counters, 0, 1024 * sizeof(unsigned), stream);
    rnnx_persistent<<<GRID, TPB, 0, stream>>>(Xt, W0, b0v, W1, b1v,
                                              out, W0h, W1h, Xh, hstates, counters);
}
